// Qwen3MoeSparseMoeBlock_36283883716952
// MI455X (gfx1250) — matrix operand order repaired and checked by reading, not compiled, not run on hardware
//
#include <hip/hip_runtime.h>
#include <stdint.h>

// Problem constants (Qwen3 MoE block)
#define T_TOK 1024
#define H_DIM 2048
#define E_NUM 64
#define I_DIM 768
#define TOPK  8
#define M_TILE 32       // token rows per workgroup tile
#define KSTEP 64        // K advance per barrier
#define NCHUNK 128      // N columns per chunk (each of 4 wave-columns owns 32)
#define TILE_E (KSTEP * NCHUNK)   // elements in one staged weight tile

typedef __attribute__((ext_vector_type(16))) __bf16 v16bf;
typedef __attribute__((ext_vector_type(8)))  float  v8f;

// ---------------------------------------------------------------- helpers
__device__ __forceinline__ uint32_t pk_bf16(float lo, float hi) {
    union { __bf16 h[2]; uint32_t u; } p;
    p.h[0] = (__bf16)lo; p.h[1] = (__bf16)hi;
    return p.u;   // v_cvt_pk_bf16_f32
}

// Load a 64(K) x 128(N) fp32 tile into registers (8 x float4 per thread).
__device__ __forceinline__ void stage_load(const float* __restrict__ src, int ld,
                                           int tid, float4* r) {
    #pragma unroll
    for (int i = 0; i < 8; ++i) {
        int f  = tid + i * 256;
        int kk = f >> 5;            // 32 float4 per 128-col row
        int c4 = (f & 31) * 4;
        r[i] = *(const float4*)(src + (size_t)kk * ld + c4);
        if (i == 0) __builtin_prefetch(src + (size_t)(kk + 2 * KSTEP) * ld + c4, 0, 1);
    }
}
// Convert + store the registers into a [64][128] bf16 LDS tile.
__device__ __forceinline__ void stage_store(__bf16* __restrict__ dst, int tid,
                                            const float4* r) {
    #pragma unroll
    for (int i = 0; i < 8; ++i) {
        int f  = tid + i * 256;
        int kk = f >> 5;
        int c4 = (f & 31) * 4;
        uint2 p;
        p.x = pk_bf16(r[i].x, r[i].y);
        p.y = pk_bf16(r[i].z, r[i].w);
        *(uint2*)(dst + kk * NCHUNK + c4) = p;
    }
}

// A fragment: 16x32 bf16, CDNA5 wave32 layout. arow = LDS row base, kbase absolute.
__device__ __forceinline__ v16bf load_a_frag(const __bf16* __restrict__ arow,
                                             int kbase, int half) {
    v16bf a;
    #pragma unroll
    for (int v = 0; v < 8; ++v) {
        int kb = kbase + ((v & 4) << 2) + half * 8 + (v & 3) * 2;
        a[2*v]   = arow[kb];
        a[2*v+1] = arow[kb + 1];
    }
    return a;
}
// B fragment: 32x16 bf16 from a [.][128] LDS tile; n col = lane%16, k in table order.
__device__ __forceinline__ v16bf load_b_frag(const __bf16* __restrict__ tile,
                                             int lane, int colbase) {
    v16bf b;
    const __bf16* p = tile + colbase + (lane & 15);
    #pragma unroll
    for (int j = 0; j < 16; ++j) b[j] = p[(((j & 8) << 1) + (lane >> 4) * 8 + (j & 7)) * NCHUNK];
    return b;
}

#define WMMA_BF16(A, B, C) \
    __builtin_amdgcn_wmma_f32_16x16x32_bf16(false, (A), false, (B), (short)0, (C), false, false)

// ---------------------------------------------------------------- utilities
__global__ void k_zero_counts(int* counts) {
    if (threadIdx.x < E_NUM) counts[threadIdx.x] = 0;
}

// ---------------------------------------------------------------- router
// One wave32 per token. Lane l owns experts l and l+32.
__global__ void k_router(const float* __restrict__ x,
                         const float* __restrict__ gate_w,
                         int* __restrict__ counts,
                         uint32_t* __restrict__ pairs,
                         float* __restrict__ wts) {
    const int wave = threadIdx.x >> 5;
    const int lane = threadIdx.x & 31;
    const int t = blockIdx.x * 4 + wave;
    const int e0 = lane, e1 = lane + 32;

    const float* xr = x + (size_t)t * H_DIM;
    const float* g0 = gate_w + (size_t)e0 * H_DIM;
    const float* g1 = gate_w + (size_t)e1 * H_DIM;
    float d0 = 0.f, d1 = 0.f;
    for (int k = 0; k < H_DIM; k += 4) {
        float4 xv = *(const float4*)(xr + k);
        float4 a  = *(const float4*)(g0 + k);
        float4 b  = *(const float4*)(g1 + k);
        d0 += xv.x*a.x + xv.y*a.y + xv.z*a.z + xv.w*a.w;
        d1 += xv.x*b.x + xv.y*b.y + xv.z*b.z + xv.w*b.w;
    }

    // 8 rounds of wave-wide argmax; lane `it` captures round `it`'s winner.
    float my_v = 0.f, m8 = 0.f; int my_e = 0;
    #pragma unroll
    for (int it = 0; it < TOPK; ++it) {
        float bv; int be;
        if (d0 >= d1) { bv = d0; be = e0; } else { bv = d1; be = e1; }
        #pragma unroll
        for (int off = 16; off >= 1; off >>= 1) {
            float ov = __shfl_xor(bv, off, 32);
            int   oe = __shfl_xor(be, off, 32);
            if (ov > bv || (ov == bv && oe < be)) { bv = ov; be = oe; }
        }
        if (it == 0) m8 = bv;                 // max of the top-8 (uniform)
        if (lane == it) { my_v = bv; my_e = be; }
        if (be == e0) d0 = -__builtin_inff();
        if (be == e1) d1 = -__builtin_inff();
    }

    // renormalized softmax weights over the top-8 (softmax monotonic => same set)
    float contrib = (lane < TOPK) ? __expf(my_v - m8) : 0.f;
    float wsum = contrib;
    #pragma unroll
    for (int off = 16; off >= 1; off >>= 1) wsum += __shfl_xor(wsum, off, 32);

    if (lane < TOPK) {
        int slot = atomicAdd(&counts[my_e], 1);
        pairs[my_e * T_TOK + slot] = (uint32_t)t | ((uint32_t)lane << 16);
        wts[my_e * T_TOK + slot] = contrib / wsum;
    }
}

// ---------------------------------------------------------------- expert MLP
// Block = 256 threads = 8 waves arranged 2(M) x 4(N); each wave owns 2 N-slices.
// Stage 1: [32, H] x [H, I] -> h = silu(g)*u  (bf16 in LDS)
// Stage 2: [32, I] x [I, H] -> row-scaled scatter to per-(t,k) scratch (deterministic)
__global__ void __launch_bounds__(256)
k_expert(const float* __restrict__ x,
         const float* __restrict__ w_gate,
         const float* __restrict__ w_up,
         const float* __restrict__ w_down,
         const int* __restrict__ counts,
         const uint32_t* __restrict__ pairs,
         const float* __restrict__ wts,
         float* __restrict__ scratch) {
    const int e   = blockIdx.x;
    const int cnt = counts[e];
    const int m0  = blockIdx.y * M_TILE;
    if (m0 >= cnt) return;

    extern __shared__ char smem_raw[];
    __bf16* xs   = (__bf16*)smem_raw;            // [32][2048]  128 KB
    __bf16* hs   = xs + M_TILE * H_DIM;          // [32][768]    48 KB
    // Double-buffered weight tiles, selected by INTEGER offset only (keeps
    // LDS addrspace inference intact -> ds_load, not flat_load):
    //   buffer i: G tile at wbuf + i*2*TILE_E, U tile at +TILE_E.
    __bf16* wbuf = hs + M_TILE * I_DIM;          // 4 * 16 KB = 64 KB
    int*   drow = (int*)(wbuf + 4 * TILE_E);     // [32] scratch row (t*8+k) or -1
    float* rw   = (float*)(drow + M_TILE);       // [32] routing weight
    int*   trow = (int*)(rw + M_TILE);           // [32] token id

    const int tid  = threadIdx.x;
    const int wave = tid >> 5;
    const int lane = tid & 31;
    const int wm   = wave & 1;     // 16-row half
    const int wn   = wave >> 1;    // 32-col slice of a 128-col chunk
    const int half = lane >> 4;
    const int l15  = lane & 15;

    if (tid < M_TILE) {
        int slot = m0 + tid;
        if (slot < cnt) {
            uint32_t p = pairs[e * T_TOK + slot];
            trow[tid] = (int)(p & 0xFFFFu);
            drow[tid] = (int)(p & 0xFFFFu) * TOPK + (int)(p >> 16);
            rw[tid]   = wts[e * T_TOK + slot];
        } else {
            trow[tid] = 0; drow[tid] = -1; rw[tid] = 0.f;
        }
    }
    __syncthreads();

    // ---- Phase A: gather x rows -> LDS (fp32 -> packed bf16)
    for (int i = tid; i < M_TILE * (H_DIM / 4); i += 256) {
        int r  = i >> 9;                 // 512 float4 per row
        int c4 = (i & 511) * 4;
        float4 v = *(const float4*)(x + (size_t)trow[r] * H_DIM + c4);
        uint2 p; p.x = pk_bf16(v.x, v.y); p.y = pk_bf16(v.z, v.w);
        *(uint2*)(xs + r * H_DIM + c4) = p;
    }

    // ---- Stage 1: g,u GEMMs + fused SwiGLU (software-pipelined weight stream)
    const float* wg = w_gate + (size_t)e * H_DIM * I_DIM;
    const float* wu = w_up   + (size_t)e * H_DIM * I_DIM;
    const __bf16* arow1 = xs + (wm * 16 + l15) * H_DIM;

    for (int nc = 0; nc < I_DIM / NCHUNK; ++nc) {          // 6 chunks
        const int cbase = nc * NCHUNK;
        const int cw = cbase + wn * 32;
        v8f ag0 = {}, ag1 = {}, au0 = {}, au1 = {};

        float4 rg[8], ru[8];
        stage_load(wg + cbase, I_DIM, tid, rg);
        stage_load(wu + cbase, I_DIM, tid, ru);
        stage_store(wbuf, tid, rg);               // buffer 0: G
        stage_store(wbuf + TILE_E, tid, ru);      // buffer 0: U

        for (int it = 0, k0 = 0; k0 < H_DIM; k0 += KSTEP, ++it) {
            __syncthreads();
            const int cur = it & 1, nxt = cur ^ 1;
            const bool more = (k0 + KSTEP) < H_DIM;
            if (more) {   // issue next tile's global loads (waited after compute)
                stage_load(wg + (size_t)(k0 + KSTEP) * I_DIM + cbase, I_DIM, tid, rg);
                stage_load(wu + (size_t)(k0 + KSTEP) * I_DIM + cbase, I_DIM, tid, ru);
            }
            const __bf16* gcur = wbuf + cur * (2 * TILE_E);
            const __bf16* ucur = gcur + TILE_E;
            #pragma unroll
            for (int ks = 0; ks < 2; ++ks) {
                v16bf a = load_a_frag(arow1, k0 + ks * 32, half);
                const __bf16* tg = gcur + ks * 32 * NCHUNK;
                const __bf16* tu = ucur + ks * 32 * NCHUNK;
                v16bf bg0 = load_b_frag(tg, lane, wn * 32);
                v16bf bg1 = load_b_frag(tg, lane, wn * 32 + 16);
                v16bf bu0 = load_b_frag(tu, lane, wn * 32);
                v16bf bu1 = load_b_frag(tu, lane, wn * 32 + 16);
                ag0 = WMMA_BF16(a, bg0, ag0);
                ag1 = WMMA_BF16(a, bg1, ag1);
                au0 = WMMA_BF16(a, bu0, au0);
                au1 = WMMA_BF16(a, bu1, au1);
            }
            if (more) {
                __bf16* gnxt = wbuf + nxt * (2 * TILE_E);
                stage_store(gnxt, tid, rg);
                stage_store(gnxt + TILE_E, tid, ru);
            }
        }
        // h = silu(g) * u -> LDS (C/D layout: lanes0-15 rows r, lanes16-31 rows r+8)
        #pragma unroll
        for (int s = 0; s < 2; ++s) {
            v8f gacc = s ? ag1 : ag0;
            v8f uacc = s ? au1 : au0;
            #pragma unroll
            for (int r = 0; r < 8; ++r) {
                float gv = gacc[r], uv = uacc[r];
                float sv = gv / (1.f + __expf(-gv));
                int row = wm * 16 + r + half * 8;
                hs[row * I_DIM + cw + s * 16 + l15] = (__bf16)(sv * uv);
            }
        }
    }

    // ---- Stage 2: out = (h @ w_down) * route_weight -> scratch[(t*8+k)][H]
    const float* wd = w_down + (size_t)e * I_DIM * H_DIM;
    const __bf16* arow2 = hs + (wm * 16 + l15) * I_DIM;

    for (int nc = 0; nc < H_DIM / NCHUNK; ++nc) {          // 16 chunks
        const int cbase = nc * NCHUNK;
        const int cw = cbase + wn * 32;
        v8f ac0 = {}, ac1 = {};

        float4 rg[8];
        stage_load(wd + cbase, H_DIM, tid, rg);
        stage_store(wbuf, tid, rg);               // buffer 0

        for (int it = 0, k0 = 0; k0 < I_DIM; k0 += KSTEP, ++it) {
            __syncthreads();
            const int cur = it & 1, nxt = cur ^ 1;
            const bool more = (k0 + KSTEP) < I_DIM;
            if (more)
                stage_load(wd + (size_t)(k0 + KSTEP) * H_DIM + cbase, H_DIM, tid, rg);
            const __bf16* dcur = wbuf + cur * (2 * TILE_E);
            #pragma unroll
            for (int ks = 0; ks < 2; ++ks) {
                v16bf a = load_a_frag(arow2, k0 + ks * 32, half);
                const __bf16* td = dcur + ks * 32 * NCHUNK;
                v16bf b0 = load_b_frag(td, lane, wn * 32);
                v16bf b1 = load_b_frag(td, lane, wn * 32 + 16);
                ac0 = WMMA_BF16(a, b0, ac0);
                ac1 = WMMA_BF16(a, b1, ac1);
            }
            if (more) stage_store(wbuf + nxt * (2 * TILE_E), tid, rg);
        }
        #pragma unroll
        for (int s = 0; s < 2; ++s) {
            v8f acc = s ? ac1 : ac0;
            #pragma unroll
            for (int r = 0; r < 8; ++r) {
                int row = wm * 16 + r + half * 8;
                int dr = drow[row];
                if (dr >= 0)
                    scratch[(size_t)dr * H_DIM + cw + s * 16 + l15] = acc[r] * rw[row];
            }
        }
    }
}

// ---------------------------------------------------------------- combine
__global__ void k_reduce(const float4* __restrict__ scratch4, float4* __restrict__ out4) {
    int idx = blockIdx.x * 256 + threadIdx.x;   // over T*H/4
    int t = idx >> 9;                           // H/4 = 512
    int h = idx & 511;
    float4 s = {0.f, 0.f, 0.f, 0.f};
    #pragma unroll
    for (int k = 0; k < TOPK; ++k) {
        float4 v = scratch4[(size_t)(t * TOPK + k) * (H_DIM / 4) + h];
        s.x += v.x; s.y += v.y; s.z += v.z; s.w += v.w;
    }
    out4[idx] = s;
}

// ---------------------------------------------------------------- launcher
extern "C" void kernel_launch(void* const* d_in, const int* in_sizes, int n_in,
                              void* d_out, int out_size, void* d_ws, size_t ws_size,
                              hipStream_t stream) {
    const float* x      = (const float*)d_in[0];   // [T, H]
    const float* gate_w = (const float*)d_in[1];   // [E, H]
    const float* w_gate = (const float*)d_in[2];   // [E, H, I]
    const float* w_up   = (const float*)d_in[3];   // [E, H, I]
    const float* w_down = (const float*)d_in[4];   // [E, I, H]

    char* ws = (char*)d_ws;
    int*      counts  = (int*)ws;                                    // 64 ints
    uint32_t* pairs   = (uint32_t*)(ws + 1024);                      // E*T
    float*    wts     = (float*)(ws + 1024 + E_NUM * T_TOK * 4);     // E*T
    float*    scratch = (float*)(ws + 1024 + 2 * E_NUM * T_TOK * 4); // T*K*H floats

    k_zero_counts<<<1, 64, 0, stream>>>(counts);
    k_router<<<T_TOK / 4, 128, 0, stream>>>(x, gate_w, counts, pairs, wts);

    size_t smem = (size_t)(M_TILE * H_DIM + M_TILE * I_DIM + 4 * TILE_E) * 2
                + (size_t)M_TILE * 12;
    k_expert<<<dim3(E_NUM, T_TOK / M_TILE), 256, smem, stream>>>(
        x, w_gate, w_up, w_down, counts, pairs, wts, scratch);

    k_reduce<<<(T_TOK * H_DIM / 4) / 256, 256, 0, stream>>>(
        (const float4*)scratch, (float4*)d_out);
}
